// E3_TransformerLayer_87136296501553
// MI455X (gfx1250) — compile-verified, matrix operand order checked by reading, not run on hardware
//
#include <hip/hip_runtime.h>
#include <hip/hip_bf16.h>

// E(3)-equivariant transformer layer for MI455X (gfx1250, wave32, WMMA).
// Pipeline: [A] node prep (q, zero z/f_new) -> [B] edge pass1 (radial MLPs via
// v_wmma_f32_16x16x32_f16, w3j tensor products, attention logits, atomic z)
// -> [C] edge pass2 (softmax-normalized scatter of v_l0) -> [D] node out MLP.
// Only f_new[:,0:10] feeds the output, so the l=1/l=2 blocks of v are dead
// code and are skipped (only 40 of 170 fcv2 columns are live).

typedef __attribute__((ext_vector_type(16))) _Float16 v16h;
typedef __attribute__((ext_vector_type(8)))  _Float16 v8h;
typedef __attribute__((ext_vector_type(8)))  float    v8f;

// ---- real Wigner-3j constants (e3nn convention, derived from orthonormal
// ---- Cartesian symmetric-traceless tensor bases) ----
#define C110 0.57735027f     // 1/sqrt(3)
#define C220 0.44721360f     // 1/sqrt(5)
#define C330 0.37796447f     // 1/sqrt(7)
#define W112P 0.31622777f    // 1/sqrt(10)
#define W112Q 0.18257419f    // 1/sqrt(30)
#define W112Q2 0.36514837f   // 2/sqrt(30)
#define G1 0.26726124f       // 1/sqrt(14)
#define G2 0.21821789f       // 1/sqrt(21)
#define G3 0.06900656f       // 1/sqrt(210)
#define G4 0.27602622f       // 4*sqrt(3)/(3*sqrt(70))
#define G5 0.23904572f       // 2/sqrt(70)
#define G6 0.16903085f       // 1/sqrt(35)
#define G7 0.29277002f       // sqrt(3/35)
#define A1F 0.40824829f      // 1/sqrt(6)   (fan for lo=1 in k-fctp)
#define ADOT 0.070710678f    // 1/sqrt(200) (fan for dot-fctp)

__device__ __forceinline__ float silu_f(float x) { return x / (1.f + expf(-x)); }

__device__ __forceinline__ v16h ldfrag(const _Float16* p) {
  // 8 f16 at p (K-chunk 0) and 8 f16 at p+16 (K-chunk 1): matches the
  // documented wave32 16-bit A/B fragment layout.
  v8h lo = *reinterpret_cast<const v8h*>(p);
  v8h hi = *reinterpret_cast<const v8h*>(p + 16);
  v16h r;
#pragma unroll
  for (int i = 0; i < 8; ++i) { r[i] = lo[i]; r[i + 8] = hi[i]; }
  return r;
}

// ---------------- Kernel A: q = [f0*wq0, f_l1 (x) wq1], zero z/f_new ----------------
__global__ void e3t_node_prep(const float* __restrict__ f,
                              const float* __restrict__ wq0,
                              const float* __restrict__ wq1,
                              float* __restrict__ q, float* __restrict__ z,
                              float* __restrict__ fnew, int N) {
  int n = blockIdx.x * blockDim.x + threadIdx.x;
  if (n >= N) return;
  float f0 = f[n * 16 + 0];
  float f1 = f[n * 16 + 1], f2 = f[n * 16 + 2], f3 = f[n * 16 + 3];
#pragma unroll
  for (int j = 0; j < 10; ++j) q[n * 40 + j] = f0 * wq0[j];
#pragma unroll
  for (int u = 0; u < 10; ++u) {
    float w = wq1[u];
    q[n * 40 + 10 + 3 * u + 0] = f1 * w;
    q[n * 40 + 10 + 3 * u + 1] = f2 * w;
    q[n * 40 + 10 + 3 * u + 2] = f3 * w;
  }
  z[n] = 0.f;
#pragma unroll
  for (int w = 0; w < 10; ++w) fnew[n * 10 + w] = 0.f;
}

// ---------------- Kernel B: edge pass 1 (WMMA MLPs + tensor products) ----------------
__global__ __launch_bounds__(64) void e3t_edge_pass1(
    const float* __restrict__ f, const float* __restrict__ pos,
    const float* __restrict__ fck1, const float* __restrict__ fck2,
    const float* __restrict__ fcv1, const float* __restrict__ fcv2,
    const float* __restrict__ wdot, const float* __restrict__ q,
    const int* __restrict__ esrc, const int* __restrict__ edst,
    float* __restrict__ z, float* __restrict__ expv, float* __restrict__ v10g) {
  const int tid = threadIdx.x;
  const int e = blockIdx.x * 64 + tid;

  __shared__ __align__(16) _Float16 s_hk[64 * 64];   // hidden (k branch), 8 KB
  __shared__ __align__(16) _Float16 s_hv[64 * 64];   // hidden (v branch)
  __shared__ __align__(16) _Float16 s_w2k[112 * 64]; // fck2^T (cols padded 100->112), *0.125
  __shared__ __align__(16) _Float16 s_w2v[48 * 64];  // live fcv2 cols {0-9,40-49,100-109,150-159}
  __shared__ float s_w1k[256];
  __shared__ float s_w1v[256];
  __shared__ _Float16 s_c[64 * 160];                 // GEMM out: [edge][0..111]=wk, [112..151]=wv
  __shared__ float s_wdot[200];

  // ---- stage weights (transposed, scale 1/sqrt(64) folded) ----
  for (int idx = tid; idx < 112 * 64; idx += 64) {
    int n = idx >> 6, kk = idx & 63;
    s_w2k[idx] = (_Float16)((n < 100 ? fck2[kk * 100 + n] : 0.f) * 0.125f);
  }
  for (int idx = tid; idx < 48 * 64; idx += 64) {
    int n = idx >> 6, kk = idx & 63;
    float v = 0.f;
    if (n < 40) {
      int g = n / 10;
      int base = (g == 0) ? 0 : (g == 1) ? 40 : (g == 2) ? 100 : 150;
      v = fcv2[kk * 170 + base + (n % 10)] * 0.125f;
    }
    s_w2v[idx] = (_Float16)v;
  }
  for (int idx = tid; idx < 256; idx += 64) { s_w1k[idx] = fck1[idx]; s_w1v[idx] = fcv1[idx]; }
  for (int idx = tid; idx < 200; idx += 64) s_wdot[idx] = wdot[idx];
  __syncthreads();

  // ---- per-edge geometry + radial basis + MLP hidden layer ----
  const int si = esrc[e], di = edst[e];
  float vx = pos[si * 3 + 0] - pos[di * 3 + 0];
  float vy = pos[si * 3 + 1] - pos[di * 3 + 1];
  float vz = pos[si * 3 + 2] - pos[di * 3 + 2];
  float len = sqrtf(vx * vx + vy * vy + vz * vz);
  float d0 = (len - 1.6f) * 0.625f, d1 = (len - 3.2f) * 0.625f;
  float d2 = (len - 4.8f) * 0.625f, d3 = (len - 6.4f) * 0.625f;
  float b0 = 1.7857143f * expf(-d0 * d0), b1 = 1.7857143f * expf(-d1 * d1);
  float b2 = 1.7857143f * expf(-d2 * d2), b3 = 1.7857143f * expf(-d3 * d3);
  float xc = 5.f * (1.f - len * 0.125f);
  float cut = (xc > 0.f) ? expf(-1.f / xc) : 0.f;
  float inv = 1.f / fmaxf(len, 1e-9f);
  float ux = vx * inv, uy = vy * inv, uz = vz * inv;

  for (int c = 0; c < 64; ++c) {
    float hk = 0.5f * (b0 * s_w1k[c] + b1 * s_w1k[64 + c] + b2 * s_w1k[128 + c] + b3 * s_w1k[192 + c]);
    float hv = 0.5f * (b0 * s_w1v[c] + b1 * s_w1v[64 + c] + b2 * s_w1v[128 + c] + b3 * s_w1v[192 + c]);
    s_hk[tid * 64 + c] = (_Float16)silu_f(hk);
    s_hv[tid * 64 + c] = (_Float16)silu_f(hv);
  }
  __syncthreads();

  // ---- GEMM: [16 edges x 64] @ [64 x Ncols] via v_wmma_f32_16x16x32_f16 ----
  {
    const int ln = tid & 31, wv_ = tid >> 5;
    const int rr = ln & 15;
    const int h8 = (ln & 16) ? 8 : 0; // K-half select (and C row offset)
#pragma unroll
    for (int st = 0; st < 2; ++st) {
      const int tb = (wv_ * 2 + st) * 16;
      {
        v16h a0 = ldfrag(&s_hk[(tb + rr) * 64 + h8]);
        v16h a1 = ldfrag(&s_hk[(tb + rr) * 64 + 32 + h8]);
#pragma unroll
        for (int t = 0; t < 7; ++t) {
          v16h bb0 = ldfrag(&s_w2k[(t * 16 + rr) * 64 + h8]);
          v16h bb1 = ldfrag(&s_w2k[(t * 16 + rr) * 64 + 32 + h8]);
          v8f c = {};
          c = __builtin_amdgcn_wmma_f32_16x16x32_f16(false, a0, false, bb0, (short)0, c, false, false);
          c = __builtin_amdgcn_wmma_f32_16x16x32_f16(false, a1, false, bb1, (short)0, c, false, false);
#pragma unroll
          for (int r = 0; r < 8; ++r)
            s_c[(tb + h8 + r) * 160 + t * 16 + rr] = (_Float16)c[r];
        }
      }
      {
        v16h a0 = ldfrag(&s_hv[(tb + rr) * 64 + h8]);
        v16h a1 = ldfrag(&s_hv[(tb + rr) * 64 + 32 + h8]);
#pragma unroll
        for (int t = 0; t < 3; ++t) {
          v16h bb0 = ldfrag(&s_w2v[(t * 16 + rr) * 64 + h8]);
          v16h bb1 = ldfrag(&s_w2v[(t * 16 + rr) * 64 + 32 + h8]);
          v8f c = {};
          c = __builtin_amdgcn_wmma_f32_16x16x32_f16(false, a0, false, bb0, (short)0, c, false, false);
          c = __builtin_amdgcn_wmma_f32_16x16x32_f16(false, a1, false, bb1, (short)0, c, false, false);
#pragma unroll
          for (int r = 0; r < 8; ++r)
            s_c[(tb + h8 + r) * 160 + 112 + t * 16 + rr] = (_Float16)c[r];
        }
      }
    }
  }
  __syncthreads();

  // ---- per-edge tensor products (all scalar registers, no private arrays) ----
  const int soff = tid * 160;
#define SCc(p) ((float)s_c[soff + (p)])
  const float* fr = f + (size_t)si * 16;
  float xf0 = fr[0], xf1 = fr[1], xf2 = fr[2], xf3 = fr[3];
  float xf4 = fr[4], xf5 = fr[5], xf6 = fr[6], xf7 = fr[7], xf8 = fr[8];
  float xf9 = fr[9], xf10 = fr[10], xf11 = fr[11], xf12 = fr[12];
  float xf13 = fr[13], xf14 = fr[14], xf15 = fr[15];

  float xx = ux * ux, yy = uy * uy, zz2 = uz * uz;
  float xs0 = 1.f;
  float xs1 = 1.7320508f * uy, xs2 = 1.7320508f * uz, xs3 = 1.7320508f * ux;
  float xs4 = 3.8729833f * ux * uy, xs5 = 3.8729833f * uy * uz;
  float xs6 = 1.1180340f * (3.f * zz2 - 1.f);
  float xs7 = 3.8729833f * ux * uz;
  float xs8 = 1.9364917f * (xx - yy);
  float xs9 = 2.0916501f * uy * (3.f * xx - yy);
  float xs10 = 10.246951f * ux * uy * uz;
  float xs11 = 1.6201852f * uy * (5.f * zz2 - 1.f);
  float xs12 = 1.3228757f * uz * (5.f * zz2 - 3.f);
  float xs13 = 1.6201852f * ux * (5.f * zz2 - 1.f);
  float xs14 = 5.1234754f * uz * (xx - yy);
  float xs15 = 2.0916501f * ux * (xx - 3.f * yy);

  // scalar-output path contractions (shared between k and v)
  float t000 = xf0 * xs0;
  float t110 = C110 * (xf1 * xs1 + xf2 * xs2 + xf3 * xs3);
  float t220 = C220 * (xf4 * xs4 + xf5 * xs5 + xf6 * xs6 + xf7 * xs7 + xf8 * xs8);
  float t330 = C330 * (xf9 * xs9 + xf10 * xs10 + xf11 * xs11 + xf12 * xs12 +
                       xf13 * xs13 + xf14 * xs14 + xf15 * xs15);
  // l=1-output contractions for k
  float ta0 = C110 * xf0 * xs1, ta1 = C110 * xf0 * xs2, ta2 = C110 * xf0 * xs3;          // (0,1,1)
  float tb0 = C110 * xf1 * xs0, tb1 = C110 * xf2 * xs0, tb2 = C110 * xf3 * xs0;          // (1,0,1)
  float tc0 = W112P * (xf3 * xs4 + xf2 * xs5) - W112Q * xf1 * xs6 - W112P * xf1 * xs8;   // (1,2,1)
  float tc1 = W112P * (xf1 * xs5 + xf3 * xs7) + W112Q2 * xf2 * xs6;
  float tc2 = W112P * (xf1 * xs4 + xf2 * xs7 + xf3 * xs8) - W112Q * xf3 * xs6;
  float td0 = W112P * (xf4 * xs3 + xf5 * xs2) - W112Q * xf6 * xs1 - W112P * xf8 * xs1;   // (2,1,1)
  float td1 = W112P * (xf5 * xs1 + xf7 * xs3) + W112Q2 * xf6 * xs2;
  float td2 = W112P * (xf4 * xs1 + xf7 * xs2 + xf8 * xs3) - W112Q * xf6 * xs3;
  float te0 = G1 * xf8 * xs9 + G2 * xf7 * xs10 + G5 * xf6 * xs11 + G3 * xf8 * xs11      // (2,3,1)
            - G6 * xf5 * xs12 - G3 * xf4 * xs13 - G2 * xf5 * xs14 - G1 * xf4 * xs15;
  float te1 = G2 * xf4 * xs10 + G4 * xf5 * xs11 + G7 * xf6 * xs12 + G4 * xf7 * xs13 + G2 * xf8 * xs14;
  float te2 = G1 * xf4 * xs9 + G2 * xf5 * xs10 - G3 * xf4 * xs11 - G6 * xf7 * xs12
            + G5 * xf6 * xs13 - G3 * xf8 * xs13 + G2 * xf7 * xs14 + G1 * xf8 * xs15;
  float tf0 = G1 * xf9 * xs8 + G2 * xf10 * xs7 + G5 * xf11 * xs6 + G3 * xf11 * xs8      // (3,2,1)
            - G6 * xf12 * xs5 - G3 * xf13 * xs4 - G2 * xf14 * xs5 - G1 * xf15 * xs4;
  float tf1 = G2 * xf10 * xs4 + G4 * xf11 * xs5 + G7 * xf12 * xs6 + G4 * xf13 * xs7 + G2 * xf14 * xs8;
  float tf2 = G1 * xf9 * xs4 + G2 * xf10 * xs5 - G3 * xf11 * xs4 - G6 * xf12 * xs7
            + G5 * xf13 * xs6 - G3 * xf13 * xs8 + G2 * xf14 * xs7 + G1 * xf15 * xs8;

  // v (l=0 block only; fan alpha = 0.5 for every contributing path)
#pragma unroll
  for (int w = 0; w < 10; ++w)
    v10g[(size_t)e * 10 + w] =
        0.5f * (SCc(112 + w) * t000 + SCc(122 + w) * t110 +
                SCc(132 + w) * t220 + SCc(142 + w) * t330);

  // attention logit: fctp(q[dst], k, wdot) with shared weights
  const float* qg = q + (size_t)di * 40;
  float lg0 = 0.f, lg1 = 0.f;
#pragma unroll
  for (int v2 = 0; v2 < 10; ++v2) {
    float k0v = 0.5f * (SCc(v2) * t000 + SCc(30 + v2) * t110 +
                        SCc(60 + v2) * t220 + SCc(90 + v2) * t330);
    float ka0 = SCc(10 + v2) * ta0 + SCc(20 + v2) * tb0 + SCc(40 + v2) * tc0 +
                SCc(50 + v2) * td0 + SCc(70 + v2) * te0 + SCc(80 + v2) * tf0;
    float ka1 = SCc(10 + v2) * ta1 + SCc(20 + v2) * tb1 + SCc(40 + v2) * tc1 +
                SCc(50 + v2) * td1 + SCc(70 + v2) * te1 + SCc(80 + v2) * tf1;
    float ka2 = SCc(10 + v2) * ta2 + SCc(20 + v2) * tb2 + SCc(40 + v2) * tc2 +
                SCc(50 + v2) * td2 + SCc(70 + v2) * te2 + SCc(80 + v2) * tf2;
#pragma unroll
    for (int u = 0; u < 10; ++u) {
      lg0 += k0v * qg[u] * s_wdot[u * 10 + v2];
      float w1 = s_wdot[100 + u * 10 + v2];
      lg1 += w1 * (qg[10 + 3 * u + 0] * ka0 + qg[10 + 3 * u + 1] * ka1 +
                   qg[10 + 3 * u + 2] * ka2);
    }
  }
  // A1 * 1/sqrt(3) folded for the l=1 path; overall 1/sqrt(200)
  float lg = ADOT * (lg0 + (A1F * C110) * lg1);
  float ev = cut * expf(lg);
  expv[e] = ev;
  atomicAdd(&z[di], ev);
#undef SCc
}

// ---------------- Kernel C: normalized scatter ----------------
__global__ void e3t_edge_pass2(const int* __restrict__ edst,
                               const float* __restrict__ expv,
                               const float* __restrict__ z,
                               const float* __restrict__ v10g,
                               float* __restrict__ fnew, int E) {
  int e = blockIdx.x * blockDim.x + threadIdx.x;
  if (e >= E) return;
  int d = edst[e];
  float zz = z[d];
  zz = (zz == 0.f) ? 1.f : zz;
  float s = sqrtf(fmaxf(expv[e] / zz, 1e-30f));
#pragma unroll
  for (int w = 0; w < 10; ++w)
    atomicAdd(&fnew[(size_t)d * 10 + w], s * v10g[(size_t)e * 10 + w]);
}

// ---------------- Kernel D: output MLP ----------------
__global__ void e3t_node_out(const float* __restrict__ fnew,
                             const float* __restrict__ wlin1,
                             const float* __restrict__ wlin2,
                             float* __restrict__ out, int N) {
  int n = blockIdx.x * blockDim.x + threadIdx.x;
  if (n >= N) return;
  float o = 0.f;
#pragma unroll
  for (int j = 0; j < 10; ++j) {
    float acc = 0.f;
#pragma unroll
    for (int i = 0; i < 10; ++i) acc += fnew[n * 10 + i] * wlin1[i * 10 + j];
    acc *= 0.31622777f; // 1/sqrt(10)
    o += silu_f(acc) * wlin2[j];
  }
  out[n] = o * 0.31622777f;
}

extern "C" void kernel_launch(void* const* d_in, const int* in_sizes, int n_in,
                              void* d_out, int out_size, void* d_ws, size_t ws_size,
                              hipStream_t stream) {
  const float* f     = (const float*)d_in[0];
  const float* pos   = (const float*)d_in[1];
  const float* wq0   = (const float*)d_in[2];
  const float* wq1   = (const float*)d_in[3];
  const float* fck1  = (const float*)d_in[4];
  const float* fck2  = (const float*)d_in[5];
  const float* fcv1  = (const float*)d_in[6];
  const float* fcv2  = (const float*)d_in[7];
  const float* wdot  = (const float*)d_in[8];
  const float* wlin1 = (const float*)d_in[9];
  const float* wlin2 = (const float*)d_in[10];
  const int* esrc    = (const int*)d_in[11];
  const int* edst    = (const int*)d_in[12];
  float* out = (float*)d_out;

  const int N = in_sizes[0] / 16;
  const int E = in_sizes[11];

  // workspace layout (floats): q[N*40] | z[N] | fnew[N*10] | expv[E] | v10[E*10]
  float* ws   = (float*)d_ws;
  float* q    = ws;
  float* z    = q + (size_t)N * 40;
  float* fnew = z + N;
  float* expv = fnew + (size_t)N * 10;
  float* v10  = expv + E;

  e3t_node_prep<<<(N + 255) / 256, 256, 0, stream>>>(f, wq0, wq1, q, z, fnew, N);
  // E is a multiple of 64 for this problem (524288)
  e3t_edge_pass1<<<E / 64, 64, 0, stream>>>(f, pos, fck1, fck2, fcv1, fcv2,
                                            wdot, q, esrc, edst, z, expv, v10);
  e3t_edge_pass2<<<(E + 255) / 256, 256, 0, stream>>>(edst, expv, z, v10, fnew, E);
  e3t_node_out<<<(N + 255) / 256, 256, 0, stream>>>(fnew, wlin1, wlin2, out, N);
}
